// MultiStepLinearRNN_17334488006797
// MI455X (gfx1250) — matrix operand order repaired and checked by reading, not compiled, not run on hardware
//
#include <hip/hip_runtime.h>
#include <hip/hip_bf16.h>
#include <math.h>

// Problem constants (fixed by the reference)
#define BB 64
#define SS 2048
#define DD 256
#define HH 1024
#define FF 128
#define NWG 32          // persistent workgroups
#define TPB 256         // 8 wave32 per WG
#define LN_EPS 1e-5f
#define RS 1288         // LDS A-panel row stride in bf16 elems (1280 + 8 pad)

typedef __bf16 bf16_t;
typedef __attribute__((ext_vector_type(16))) bf16_t v16bf;
typedef __attribute__((ext_vector_type(8)))  bf16_t v8bf;
typedef __attribute__((ext_vector_type(8)))  float  v8f;

union ABu { v16bf v; v8bf h[2]; };

__device__ __forceinline__ unsigned short f2bf(float f) {
  unsigned u = __float_as_uint(f);
  u += 0x7FFFu + ((u >> 16) & 1u);       // round-to-nearest-even
  return (unsigned short)(u >> 16);
}

__device__ __forceinline__ float gelu_exact(float x) {
  return 0.5f * x * (1.0f + erff(x * 0.7071067811865475f));
}

// One 16x16x32 bf16 WMMA step. A fragment from LDS panel (rows 0..15,
// per-lane layout: m = lane&15, K halves selected by lane>>4), B fragment
// pre-swizzled in global (per-lane contiguous 16 bf16).
__device__ __forceinline__ v8f wmma_chunk(v8f acc, const unsigned short* sA, int rs, int kk,
                                          const unsigned short* __restrict__ wf,
                                          int nTiles, int nt, int cl, int lane) {
  ABu a;
  const unsigned short* ap = sA + (lane & 15) * rs + kk + ((lane >> 4) << 4);
  a.h[0] = *(const v8bf*)ap;        // K = kb .. kb+7
  a.h[1] = *(const v8bf*)(ap + 8);  // K = kb+8 .. kb+15
  v16bf b = *(const v16bf*)(wf + ((size_t)(cl * nTiles + nt) * 32 + lane) * 16);
  return __builtin_amdgcn_wmma_f32_16x16x32_bf16(false, a.v, false, b, (short)0, acc, false, false);
}

// Monotonic grid-wide barrier (persistent kernel, NWG resident blocks).
__device__ __forceinline__ void gridbar(unsigned* cnt, unsigned& target) {
  __syncthreads();
  target += NWG;
  if (threadIdx.x == 0) {
    __threadfence();
    atomicAdd(cnt, 1u);
    while (atomicAdd(cnt, 0u) < target) __builtin_amdgcn_s_sleep(8);
    __threadfence();
  }
  __syncthreads();
}

// ---------------- init: zero barrier counter + h states ----------------
__global__ void k_init(unsigned* cnt, float* h0, float* h1) {
  size_t idx = (size_t)blockIdx.x * TPB + threadIdx.x;
  if (idx == 0) *cnt = 0u;
  if (idx < (size_t)2 * BB * HH) { h0[idx] = 0.f; h1[idx] = 0.f; }
}

// -------- convert f32 KxN weight to bf16 per-lane B-fragment layout -----
// frag[((kt*NT+nt)*32 + lane)*16 + v] = bf16(W[k][n]),
//   n = nt*16 + (lane&15), k = kt*32 + (lane>>4)*16 + v
__global__ void k_convert(const float* __restrict__ W, unsigned short* __restrict__ frag,
                          int K, int N) {
  int idx = blockIdx.x * TPB + threadIdx.x;
  int NT = N >> 4;
  int tile = idx >> 9;
  int within = idx & 511;
  int lane = within >> 4;
  int v = within & 15;
  int kt = tile / NT;
  int nt = tile - kt * NT;
  int n = (nt << 4) + (lane & 15);
  int k = (kt << 5) + ((lane >> 4) << 4) + v;
  frag[idx] = f2bf(W[(size_t)k * N + n]);
}

// ------- u = LayerNorm(GELU(x @ stack_W + b)) for all B*S rows (bf16) ---
__global__ __launch_bounds__(TPB) void k_input_stack(
    const float* __restrict__ x, const unsigned short* __restrict__ stackWf,
    const float* __restrict__ stack_b, const float* __restrict__ ln_g,
    const float* __restrict__ ln_b, unsigned short* __restrict__ u) {
  __shared__ __align__(16) unsigned short pan[16 * 264];
  __shared__ float ybuf[16][DD];
  const int tid = threadIdx.x, lane = tid & 31, wv = tid >> 5;
  const size_t row0 = (size_t)blockIdx.x * 16;

  for (int i = tid; i < 16 * DD; i += TPB) {
    int r = i >> 8, k = i & 255;
    pan[r * 264 + k] = f2bf(x[(row0 + r) * DD + k]);
  }
  __syncthreads();

  for (int s = 0; s < 2; ++s) {            // 8 waves x 2 tiles = 16 N-tiles
    int nt = wv * 2 + s;
    v8f acc = {0.f,0.f,0.f,0.f,0.f,0.f,0.f,0.f};
    for (int c = 0; c < 8; ++c)
      acc = wmma_chunk(acc, pan, 264, c * 32, stackWf, 16, nt, c, lane);
    int n = (nt << 4) + (lane & 15);
    float bias = stack_b[n];
    for (int r = 0; r < 8; ++r)
      ybuf[r + ((lane >> 4) << 3)][n] = gelu_exact(acc[r] + bias);
  }
  __syncthreads();

  for (int s = 0; s < 2; ++s) {            // LN over 256 cols, 2 rows/wave
    int r = wv * 2 + s;
    float sum = 0.f, sq = 0.f, vals[8];
    for (int j = 0; j < 8; ++j) {
      float yv = ybuf[r][lane * 8 + j];
      vals[j] = yv; sum += yv; sq += yv * yv;
    }
    for (int off = 16; off > 0; off >>= 1) {
      sum += __shfl_xor(sum, off, 32);
      sq  += __shfl_xor(sq,  off, 32);
    }
    float mu = sum * (1.f / DD);
    float var = sq * (1.f / DD) - mu * mu;
    float inv = rsqrtf(var + LN_EPS);
    for (int j = 0; j < 8; ++j) {
      int col = lane * 8 + j;
      u[(row0 + r) * DD + col] = f2bf(ln_g[col] * (vals[j] - mu) * inv + ln_b[col]);
    }
  }
}

// --------------- persistent scan kernel: encoder + decoder --------------
__global__ __launch_bounds__(TPB) void k_rnn(
    const float* __restrict__ x, const unsigned short* __restrict__ uenc,
    const unsigned short* __restrict__ stackWf,
    const unsigned short* __restrict__ wx0f, const unsigned short* __restrict__ wh0f,
    const unsigned short* __restrict__ wx1f, const unsigned short* __restrict__ wh1f,
    const unsigned short* __restrict__ inpf, const unsigned short* __restrict__ prjf,
    const float* __restrict__ stack_b, const float* __restrict__ ln_g,
    const float* __restrict__ ln_b, const float* __restrict__ b0,
    const float* __restrict__ b1, const float* __restrict__ inproj_b,
    const float* __restrict__ proj_b, const float* __restrict__ dec0p,
    const float* __restrict__ dec1p, const float* __restrict__ skwp,
    float* __restrict__ h0buf, float* __restrict__ h1buf,
    float* __restrict__ ydec, unsigned short* __restrict__ udec,
    float* __restrict__ curbuf, unsigned* __restrict__ cnt,
    float* __restrict__ out) {
  __shared__ __align__(16) unsigned short pan[16 * RS];   // ~40 KB
  const int tid = threadIdx.x, lane = tid & 31, wv = tid >> 5;
  const int wg = blockIdx.x, mt = wg >> 3, grp = wg & 7;
  const float d0 = dec0p[0], d1 = dec1p[0], skw = skwp[0];
  unsigned target = 0;

  // stage 16 x 1024 f32 rows (scaled) into panel at column offset coff
  auto stageH = [&](const float* src, float scale, int coff) {
    for (int i = tid; i < 16 * HH; i += TPB) {
      int r = i >> 10, k = i & (HH - 1);
      pan[r * RS + coff + k] = f2bf(scale * src[(size_t)(mt * 16 + r) * HH + k]);
    }
  };
  // stage 16 x 256 f32 rows (stride mstr) scaled
  auto stageDf = [&](const float* src, size_t mstr, float scale, int coff) {
    for (int i = tid; i < 16 * DD; i += TPB) {
      int r = i >> 8, k = i & 255;
      pan[r * RS + coff + k] = f2bf(scale * src[(size_t)(mt * 16 + r) * mstr + k]);
    }
  };
  // stage 16 x 256 bf16 rows (stride mstr)
  auto stageDu = [&](const unsigned short* src, size_t mstr, int coff) {
    for (int i = tid; i < 16 * DD; i += TPB) {
      int r = i >> 8, k = i & 255;
      pan[r * RS + coff + k] = src[(size_t)(mt * 16 + r) * mstr + k];
    }
  };
  // h-state GEMM: K=1280 = [1024 from wfA | 256 from wfB], N=1024, all 8 waves
  auto gemmH = [&](const unsigned short* wfA, const unsigned short* wfB,
                   const float* bias, float* dst) {
    int nt = grp * 8 + wv;
    v8f acc = {0.f,0.f,0.f,0.f,0.f,0.f,0.f,0.f};
    for (int c = 0; c < 32; ++c)
      acc = wmma_chunk(acc, pan, RS, c * 32, wfA, 64, nt, c, lane);
    for (int c = 0; c < 8; ++c)
      acc = wmma_chunk(acc, pan, RS, (32 + c) * 32, wfB, 64, nt, c, lane);
    int n = (nt << 4) + (lane & 15);
    float bv = bias[n];
    for (int r = 0; r < 8; ++r) {
      int row = mt * 16 + r + ((lane >> 4) << 3);
      dst[(size_t)row * HH + n] = acc[r] + bv;
    }
  };
  // h1 update split in two K=1024 passes (panel reuse): acc carries over
  auto phaseH1 = [&](const float* h1r, const float* h0w, float* h1w) {
    stageH(h1r, d1, 0);
    __syncthreads();
    int nt = grp * 8 + wv;
    v8f acc = {0.f,0.f,0.f,0.f,0.f,0.f,0.f,0.f};
    for (int c = 0; c < 32; ++c)
      acc = wmma_chunk(acc, pan, RS, c * 32, wh1f, 64, nt, c, lane);
    __syncthreads();
    stageH(h0w, 1.0f, 0);
    __syncthreads();
    for (int c = 0; c < 32; ++c)
      acc = wmma_chunk(acc, pan, RS, c * 32, wx1f, 64, nt, c, lane);
    int n = (nt << 4) + (lane & 15);
    float bv = b1[n];
    for (int r = 0; r < 8; ++r) {
      int row = mt * 16 + r + ((lane >> 4) << 3);
      h1w[(size_t)row * HH + n] = acc[r] + bv;
    }
  };
  // out-projection: [h1 | skw*cur] @ [projW | inprojW], N=256, waves 0..1
  auto gemmProj = [&](int step, float* curw) {
    if (wv < 2) {
      int nt = grp * 2 + wv;   // 0..15
      v8f acc = {0.f,0.f,0.f,0.f,0.f,0.f,0.f,0.f};
      for (int c = 0; c < 32; ++c)
        acc = wmma_chunk(acc, pan, RS, c * 32, prjf, 16, nt, c, lane);
      for (int c = 0; c < 8; ++c)
        acc = wmma_chunk(acc, pan, RS, (32 + c) * 32, inpf, 16, nt, c, lane);
      int n = (nt << 4) + (lane & 15);
      float bv = proj_b[n] + skw * inproj_b[n];
      for (int r = 0; r < 8; ++r) {
        int row = mt * 16 + r + ((lane >> 4) << 3);
        float val = acc[r] + bv;
        out[(size_t)row * FF * DD + (size_t)step * DD + n] = val;
        curw[(size_t)row * DD + n] = val;
      }
    }
  };

  // ===================== encoder scan over S steps =====================
  for (int t = 0; t < SS; ++t) {
    const float* h0r = h0buf + (size_t)(t & 1) * BB * HH;
    float* h0w = h0buf + (size_t)((t + 1) & 1) * BB * HH;
    const float* h1r = h1buf + (size_t)(t & 1) * BB * HH;
    float* h1w = h1buf + (size_t)((t + 1) & 1) * BB * HH;

    // h0 = [d0*h0 | u_t] @ [Wh0 | Wx0] + b0
    stageH(h0r, d0, 0);
    stageDu(uenc + (size_t)t * DD, (size_t)SS * DD, HH);
    __syncthreads();
    gemmH(wh0f, wx0f, b0, h0w);
    gridbar(cnt, target);

    // h1 = [d1*h1 | h0_new] @ [Wh1 | Wx1] + b1
    phaseH1(h1r, h0w, h1w);
    gridbar(cnt, target);
  }
  // final states at parity 0 (S even)
  const float* h0fin = h0buf;
  const float* h1fin = h1buf;

  // ================ first prediction from x[:, S-1, :] =================
  stageH(h1fin, 1.0f, 0);
  stageDf(x + (size_t)(SS - 1) * DD, (size_t)SS * DD, skw, HH);
  __syncthreads();
  gemmProj(0, curbuf + (size_t)1 * BB * DD);   // cur[1]
  gridbar(cnt, target);
  (void)h0fin;

  // ======================= decoder: F-1 steps ==========================
  for (int s = 1; s < FF; ++s) {
    const float* curp = curbuf + (size_t)(s & 1) * BB * DD;
    float* curw = curbuf + (size_t)((s + 1) & 1) * BB * DD;
    int ph = (s - 1) & 1;
    const float* h0r = h0buf + (size_t)ph * BB * HH;
    float* h0w = h0buf + (size_t)(ph ^ 1) * BB * HH;
    const float* h1r = h1buf + (size_t)ph * BB * HH;
    float* h1w = h1buf + (size_t)(ph ^ 1) * BB * HH;

    // P0: y = GELU(cur @ stack_W + b)
    stageDf(curp, DD, 1.0f, 0);
    __syncthreads();
    if (wv < 2) {
      int nt = grp * 2 + wv;
      v8f acc = {0.f,0.f,0.f,0.f,0.f,0.f,0.f,0.f};
      for (int c = 0; c < 8; ++c)
        acc = wmma_chunk(acc, pan, RS, c * 32, stackWf, 16, nt, c, lane);
      int n = (nt << 4) + (lane & 15);
      float bias = stack_b[n];
      for (int r = 0; r < 8; ++r) {
        int row = mt * 16 + r + ((lane >> 4) << 3);
        ydec[(size_t)row * DD + n] = gelu_exact(acc[r] + bias);
      }
    }
    gridbar(cnt, target);

    // LN pass: 64 rows, one wave each (first 64 waves of the grid)
    {
      int gw = wg * 8 + wv;
      if (gw < BB) {
        float sum = 0.f, sq = 0.f, vals[8];
        for (int j = 0; j < 8; ++j) {
          float yv = ydec[(size_t)gw * DD + lane * 8 + j];
          vals[j] = yv; sum += yv; sq += yv * yv;
        }
        for (int off = 16; off > 0; off >>= 1) {
          sum += __shfl_xor(sum, off, 32);
          sq  += __shfl_xor(sq,  off, 32);
        }
        float mu = sum * (1.f / DD);
        float var = sq * (1.f / DD) - mu * mu;
        float inv = rsqrtf(var + LN_EPS);
        for (int j = 0; j < 8; ++j) {
          int col = lane * 8 + j;
          udec[(size_t)gw * DD + col] = f2bf(ln_g[col] * (vals[j] - mu) * inv + ln_b[col]);
        }
      }
    }
    gridbar(cnt, target);

    // P1: h0 update with u_dec
    stageH(h0r, d0, 0);
    stageDu(udec, DD, HH);
    __syncthreads();
    gemmH(wh0f, wx0f, b0, h0w);
    gridbar(cnt, target);

    // P2: h1 update
    phaseH1(h1r, h0w, h1w);
    gridbar(cnt, target);

    // P3: prediction
    stageH(h1w, 1.0f, 0);
    stageDf(curp, DD, skw, HH);
    __syncthreads();
    gemmProj(s, curw);
    gridbar(cnt, target);
  }
}

extern "C" void kernel_launch(void* const* d_in, const int* in_sizes, int n_in,
                              void* d_out, int out_size, void* d_ws, size_t ws_size,
                              hipStream_t stream) {
  const float* x        = (const float*)d_in[0];
  const float* stack_W  = (const float*)d_in[1];
  const float* stack_b  = (const float*)d_in[2];
  const float* ln_g     = (const float*)d_in[3];
  const float* ln_b     = (const float*)d_in[4];
  const float* Wx0      = (const float*)d_in[5];
  const float* b0       = (const float*)d_in[6];
  const float* Wh0      = (const float*)d_in[7];
  const float* decay0   = (const float*)d_in[8];
  const float* Wx1      = (const float*)d_in[9];
  const float* b1       = (const float*)d_in[10];
  const float* Wh1      = (const float*)d_in[11];
  const float* decay1   = (const float*)d_in[12];
  const float* inproj_W = (const float*)d_in[13];
  const float* inproj_b = (const float*)d_in[14];
  const float* skip_w   = (const float*)d_in[15];
  const float* proj_W   = (const float*)d_in[16];
  const float* proj_b   = (const float*)d_in[17];
  float* out = (float*)d_out;

  char* ws = (char*)d_ws;
  size_t o = 0;
  auto take = [&](size_t bytes) {
    char* p = ws + o;
    o += (bytes + 255) & ~(size_t)255;
    return p;
  };
  unsigned*       cnt     = (unsigned*)take(256);
  unsigned short* stackWf = (unsigned short*)take((size_t)DD * DD * 2);
  unsigned short* wx0f    = (unsigned short*)take((size_t)DD * HH * 2);
  unsigned short* wh0f    = (unsigned short*)take((size_t)HH * HH * 2);
  unsigned short* wx1f    = (unsigned short*)take((size_t)HH * HH * 2);
  unsigned short* wh1f    = (unsigned short*)take((size_t)HH * HH * 2);
  unsigned short* inpf    = (unsigned short*)take((size_t)DD * DD * 2);
  unsigned short* prjf    = (unsigned short*)take((size_t)HH * DD * 2);
  unsigned short* uenc    = (unsigned short*)take((size_t)BB * SS * DD * 2);
  float*          h0buf   = (float*)take((size_t)2 * BB * HH * 4);
  float*          h1buf   = (float*)take((size_t)2 * BB * HH * 4);
  float*          ydec    = (float*)take((size_t)BB * DD * 4);
  unsigned short* udec    = (unsigned short*)take((size_t)BB * DD * 2);
  float*          curbuf  = (float*)take((size_t)2 * BB * DD * 4);

  k_init<<<(2 * BB * HH + TPB - 1) / TPB, TPB, 0, stream>>>(cnt, h0buf, h1buf);

  k_convert<<<(DD * DD) / TPB, TPB, 0, stream>>>(stack_W,  stackWf, DD, DD);
  k_convert<<<(DD * HH) / TPB, TPB, 0, stream>>>(Wx0,      wx0f,    DD, HH);
  k_convert<<<(HH * HH) / TPB, TPB, 0, stream>>>(Wh0,      wh0f,    HH, HH);
  k_convert<<<(HH * HH) / TPB, TPB, 0, stream>>>(Wx1,      wx1f,    HH, HH);
  k_convert<<<(HH * HH) / TPB, TPB, 0, stream>>>(Wh1,      wh1f,    HH, HH);
  k_convert<<<(DD * DD) / TPB, TPB, 0, stream>>>(inproj_W, inpf,    DD, DD);
  k_convert<<<(HH * DD) / TPB, TPB, 0, stream>>>(proj_W,   prjf,    HH, DD);

  k_input_stack<<<(BB * SS) / 16, TPB, 0, stream>>>(x, stackWf, stack_b, ln_g, ln_b, uenc);

  k_rnn<<<NWG, TPB, 0, stream>>>(x, uenc, stackWf, wx0f, wh0f, wx1f, wh1f, inpf, prjf,
                                 stack_b, ln_g, ln_b, b0, b1, inproj_b, proj_b,
                                 decay0, decay1, skip_w,
                                 h0buf, h1buf, ydec, udec, curbuf, cnt, out);
}
